// HierarchicalHeteroGraphSage_87033217286443
// MI455X (gfx1250) — matrix operand order repaired and checked by reading, not compiled, not run on hardware
//
#include <hip/hip_runtime.h>

#define N_NODES 100000
#define E_EDGES 500000
#define T_REL   4
#define D_IN    128
#define D_H     256
#define D_OUT   64

typedef __bf16 bf16_t;
typedef bf16_t v16bf __attribute__((ext_vector_type(16)));
typedef bf16_t v8bf  __attribute__((ext_vector_type(8)));
typedef float  v8f   __attribute__((ext_vector_type(8)));

__device__ __forceinline__ bf16_t f2bf(float f) {
    union { float f; unsigned u; } in; in.f = f;
    unsigned u = in.u;
    u += 0x7fffu + ((u >> 16) & 1u);          // round-to-nearest-even
    union { unsigned short s; bf16_t b; } out;
    out.s = (unsigned short)(u >> 16);
    return out.b;
}
__device__ __forceinline__ float bf2f(bf16_t b) {
    union { bf16_t b; unsigned short s; } in; in.b = b;
    union { unsigned u; float f; } out;
    out.u = ((unsigned)in.s) << 16;
    return out.f;
}

// ---------------- aggregation kernels ----------------

__global__ void count_kernel(const int* __restrict__ dst, float* __restrict__ cnt, int E) {
    int e = blockIdx.x * blockDim.x + threadIdx.x;
    if (e < E) atomicAdd(&cnt[dst[e]], 1.0f);
}

__global__ void inv_kernel(const float* __restrict__ cnt, float* __restrict__ rinv, int n) {
    int i = blockIdx.x * blockDim.x + threadIdx.x;
    if (i < n) rinv[i] = 1.0f / fmaxf(cnt[i], 1.0f);
}

// scatter-add h[src] rows into agg[dst] rows; one thread per float4 chunk
__global__ void scatter_kernel(const float* __restrict__ h, const int* __restrict__ src,
                               const int* __restrict__ dst, float* __restrict__ agg,
                               int chunkShift, int E) {
    int tid = blockIdx.x * blockDim.x + threadIdx.x;
    int e = tid >> chunkShift;
    if (e >= E) return;
    int c = tid & ((1 << chunkShift) - 1);
    int d = 4 << chunkShift;
    int s = src[e], t = dst[e];
    const float4 v = *((const float4*)(h + (size_t)s * d) + c);
    float* p = agg + (size_t)t * d + (size_t)c * 4;
    atomicAdd(p + 0, v.x);
    atomicAdd(p + 1, v.y);
    atomicAdd(p + 2, v.z);
    atomicAdd(p + 3, v.w);
}

// ---------------- weight prep (transpose + split-bf16) ----------------

__global__ void wrsum_t_kernel(const float* __restrict__ Wr, bf16_t* __restrict__ Wt_hi,
                               bf16_t* __restrict__ Wt_lo, int K, int H) {
    int i = blockIdx.x * blockDim.x + threadIdx.x;
    if (i >= K * H) return;
    int k = i / H, n = i % H;
    float s = 0.f;
    for (int t = 0; t < T_REL; ++t) s += Wr[((size_t)t * K + k) * H + n];
    bf16_t hi = f2bf(s);
    Wt_hi[(size_t)n * K + k] = hi;
    Wt_lo[(size_t)n * K + k] = f2bf(s - bf2f(hi));
}

__global__ void bsum_kernel(const float* __restrict__ bl, float* __restrict__ bsum, int H) {
    int n = blockIdx.x * blockDim.x + threadIdx.x;
    if (n >= H) return;
    float s = 0.f;
    for (int t = 0; t < T_REL; ++t) s += bl[t * H + n];
    bsum[n] = s;
}

__global__ void wt_kernel(const float* __restrict__ W, bf16_t* __restrict__ Wt_hi,
                          bf16_t* __restrict__ Wt_lo, int K, int H) {
    int i = blockIdx.x * blockDim.x + threadIdx.x;
    if (i >= K * H) return;
    int k = i / H, n = i % H;
    float w = W[(size_t)k * H + n];
    bf16_t hi = f2bf(w);
    Wt_hi[(size_t)n * K + k] = hi;
    Wt_lo[(size_t)n * K + k] = f2bf(w - bf2f(hi));
}

// ---------------- WMMA GEMM ----------------
// C[m,n] (+)= sum_k (A[m,k]*scale_m) * W[k,n], split-bf16 (3 WMMAs / tile / k-step).
// Block: 128 threads = 4 waves; block tile 16 rows x (4*NT*16) cols.
// Weights pre-transposed bf16: Wt[n*K + k]. LDS A tile double-buffered.

__device__ __forceinline__ void stage_tile(const float* __restrict__ A, float sc,
                                           int m0, int srow, int scol, int K, int kb,
                                           bf16_t* __restrict__ ph_base,
                                           bf16_t* __restrict__ pl_base) {
    const float4 v = *(const float4*)(A + (size_t)(m0 + srow) * K + kb + scol);
    float f[4] = { v.x * sc, v.y * sc, v.z * sc, v.w * sc };
    bf16_t* ph = ph_base + srow * 40 + scol;
    bf16_t* pl = pl_base + srow * 40 + scol;
#pragma unroll
    for (int q = 0; q < 4; ++q) {
        bf16_t hi = f2bf(f[q]);
        ph[q] = hi;
        pl[q] = f2bf(f[q] - bf2f(hi));
    }
}

template<int NT>
__global__ void gemm_wmma_kernel(const float* __restrict__ A, const float* __restrict__ rinv,
                                 const bf16_t* __restrict__ Wt_hi, const bf16_t* __restrict__ Wt_lo,
                                 const float* __restrict__ bias, float* __restrict__ C,
                                 int K, int Hc, int accumulateC, int fuseRelu) {
    __shared__ bf16_t As_hi[2][16 * 40];   // stride 40 -> conflict-free b128 reads
    __shared__ bf16_t As_lo[2][16 * 40];

    const int tid  = threadIdx.x;
    const int wave = tid >> 5;
    const int lane = tid & 31;
    const int g    = lane >> 4;     // lane half
    const int ln   = lane & 15;
    const int m0   = blockIdx.x * 16;
    const int nbase = blockIdx.y * (NT * 64);

    v8f acc[NT];
#pragma unroll
    for (int j = 0; j < NT; ++j) {
        const int n0 = nbase + (wave * NT + j) * 16;
        if (accumulateC) {
#pragma unroll
            for (int v = 0; v < 8; ++v)
                acc[j][v] = C[(size_t)(m0 + v + 8 * g) * Hc + n0 + ln];
        } else if (bias) {
            float b = bias[n0 + ln];
#pragma unroll
            for (int v = 0; v < 8; ++v) acc[j][v] = b;
        } else {
#pragma unroll
            for (int v = 0; v < 8; ++v) acc[j][v] = 0.f;
        }
    }

    const int srow = tid >> 3;        // 0..15
    const int scol = (tid & 7) * 4;   // 0,4,...,28
    const float sc = rinv ? rinv[m0 + srow] : 1.0f;   // hoisted mean scale

    stage_tile(A, sc, m0, srow, scol, K, 0, As_hi[0], As_lo[0]);
    __syncthreads();

    for (int kb = 0; kb < K; kb += 32) {
        const int cur = (kb >> 5) & 1;
        if (kb + 32 < K)    // prefetch next tile into the other buffer
            stage_tile(A, sc, m0, srow, scol, K, kb + 32, As_hi[cur ^ 1], As_lo[cur ^ 1]);

        // A fragments in contiguous k order (same as B):
        // lane (ln,g): row M=ln, K runs [16g..16g+15]
        v16bf a_hi, a_lo;
        {
            const v8bf h0 = *(const v8bf*)&As_hi[cur][ln * 40 + 16 * g];
            const v8bf h1 = *(const v8bf*)&As_hi[cur][ln * 40 + 16 * g + 8];
            const v8bf l0 = *(const v8bf*)&As_lo[cur][ln * 40 + 16 * g];
            const v8bf l1 = *(const v8bf*)&As_lo[cur][ln * 40 + 16 * g + 8];
#pragma unroll
            for (int i = 0; i < 8; ++i) {
                a_hi[i] = h0[i]; a_hi[i + 8] = h1[i];
                a_lo[i] = l0[i]; a_lo[i + 8] = l1[i];
            }
        }

        // load ALL B fragments first, then interleave WMMA chains across
        // accumulators so the bf16 WMMA->WMMA hazard is hidden by independent work
        v16bf b_hi[NT], b_lo[NT];
#pragma unroll
        for (int j = 0; j < NT; ++j) {
            const int n0 = nbase + (wave * NT + j) * 16;
            // B layout: lane (ln,g): col N=ln, K = kb + 16g + e
            const bf16_t* bph = Wt_hi + (size_t)(n0 + ln) * K + kb + 16 * g;
            const bf16_t* bpl = Wt_lo + (size_t)(n0 + ln) * K + kb + 16 * g;
            const v8bf bh0 = *(const v8bf*)bph;
            const v8bf bh1 = *(const v8bf*)(bph + 8);
            const v8bf bl0 = *(const v8bf*)bpl;
            const v8bf bl1 = *(const v8bf*)(bpl + 8);
#pragma unroll
            for (int i = 0; i < 8; ++i) {
                b_hi[j][i] = bh0[i]; b_hi[j][i + 8] = bh1[i];
                b_lo[j][i] = bl0[i]; b_lo[j][i + 8] = bl1[i];
            }
        }
#pragma unroll
        for (int j = 0; j < NT; ++j)
            acc[j] = __builtin_amdgcn_wmma_f32_16x16x32_bf16(false, a_lo, false, b_hi[j],
                                                             (short)0, acc[j], false, false);
#pragma unroll
        for (int j = 0; j < NT; ++j)
            acc[j] = __builtin_amdgcn_wmma_f32_16x16x32_bf16(false, a_hi, false, b_lo[j],
                                                             (short)0, acc[j], false, false);
#pragma unroll
        for (int j = 0; j < NT; ++j)
            acc[j] = __builtin_amdgcn_wmma_f32_16x16x32_bf16(false, a_hi, false, b_hi[j],
                                                             (short)0, acc[j], false, false);

        __syncthreads();   // staged tile ready / current tile fully consumed
    }

#pragma unroll
    for (int j = 0; j < NT; ++j) {
        const int n0 = nbase + (wave * NT + j) * 16;
#pragma unroll
        for (int v = 0; v < 8; ++v) {
            float r = acc[j][v];
            if (fuseRelu) r = fmaxf(r, 0.f);
            C[(size_t)(m0 + v + 8 * g) * Hc + n0 + ln] = r;
        }
    }
}

// ---------------- launcher ----------------

extern "C" void kernel_launch(void* const* d_in, const int* in_sizes, int n_in,
                              void* d_out, int out_size, void* d_ws, size_t ws_size,
                              hipStream_t stream) {
    const float* x    = (const float*)d_in[0];
    const int*   ei[T_REL] = { (const int*)d_in[1], (const int*)d_in[2],
                               (const int*)d_in[3], (const int*)d_in[4] };
    const float* Wl0  = (const float*)d_in[5];
    const float* Wr0  = (const float*)d_in[6];
    const float* bl0  = (const float*)d_in[7];
    const float* Wl   = (const float*)d_in[8];
    const float* Wr   = (const float*)d_in[9];
    const float* bl   = (const float*)d_in[10];
    const float* Wout = (const float*)d_in[11];
    const float* bout = (const float*)d_in[12];

    char* ws = (char*)d_ws;
    size_t off = 0;
    float* h_a  = (float*)(ws + off); off += (size_t)N_NODES * D_H * 4;  // 102.4 MB
    float* h_b  = (float*)(ws + off); off += (size_t)N_NODES * D_H * 4;  // 102.4 MB
    float* agg  = (float*)(ws + off); off += (size_t)N_NODES * D_H * 4;  // 102.4 MB
    float* cnt  = (float*)(ws + off); off += (size_t)T_REL * N_NODES * 4;
    float* rinv = (float*)(ws + off); off += (size_t)T_REL * N_NODES * 4;
    float* bsum = (float*)(ws + off); off += 1024;
    bf16_t* wt_hi[5]; bf16_t* wt_lo[5];
    for (int i = 0; i < 5; ++i) {
        wt_hi[i] = (bf16_t*)(ws + off); off += (size_t)D_H * D_H * 2;
        wt_lo[i] = (bf16_t*)(ws + off); off += (size_t)D_H * D_H * 2;
    }

    // --- per-relation in-degree counts (fixed across layers) ---
    hipMemsetAsync(cnt, 0, (size_t)T_REL * N_NODES * 4, stream);
    for (int t = 0; t < T_REL; ++t)
        count_kernel<<<(E_EDGES + 255) / 256, 256, 0, stream>>>(
            ei[t] + E_EDGES, cnt + (size_t)t * N_NODES, E_EDGES);
    inv_kernel<<<(T_REL * N_NODES + 255) / 256, 256, 0, stream>>>(cnt, rinv, T_REL * N_NODES);

    const float* cur = x;
    float* bufs[2] = { h_a, h_b };

    for (int layer = 0; layer < 3; ++layer) {
        const int K = (layer == 0) ? D_IN : D_H;
        const float* WlL = (layer == 0) ? Wl0 : Wl + (size_t)(layer - 1) * T_REL * D_H * D_H;
        const float* WrL = (layer == 0) ? Wr0 : Wr + (size_t)(layer - 1) * T_REL * D_H * D_H;
        const float* blL = (layer == 0) ? bl0 : bl + (size_t)(layer - 1) * T_REL * D_H;
        float* out = bufs[layer & 1];   // L0->h_a, L1->h_b, L2->h_a

        // weight prep: summed Wr (transposed split-bf16) + summed bias + per-rel Wl
        wrsum_t_kernel<<<(K * D_H + 255) / 256, 256, 0, stream>>>(WrL, wt_hi[4], wt_lo[4], K, D_H);
        bsum_kernel<<<1, D_H, 0, stream>>>(blL, bsum, D_H);
        for (int t = 0; t < T_REL; ++t)
            wt_kernel<<<(K * D_H + 255) / 256, 256, 0, stream>>>(
                WlL + (size_t)t * K * D_H, wt_hi[t], wt_lo[t], K, D_H);

        // self term: out = cur @ Wr_sum + b_sum
        dim3 grid(N_NODES / 16, 1);
        gemm_wmma_kernel<4><<<grid, 128, 0, stream>>>(
            cur, nullptr, wt_hi[4], wt_lo[4], bsum, out, K, D_H, 0, 0);

        // per-relation mean terms, accumulated into out (ReLU fused into last)
        const int chunkShift = (K == 128) ? 5 : 6;
        const int nThreads = E_EDGES << chunkShift;
        for (int t = 0; t < T_REL; ++t) {
            hipMemsetAsync(agg, 0, (size_t)N_NODES * K * 4, stream);
            scatter_kernel<<<(nThreads + 255) / 256, 256, 0, stream>>>(
                cur, ei[t], ei[t] + E_EDGES, agg, chunkShift, E_EDGES);
            gemm_wmma_kernel<4><<<grid, 128, 0, stream>>>(
                agg, rinv + (size_t)t * N_NODES, wt_hi[t], wt_lo[t], nullptr, out,
                K, D_H, 1, (t == T_REL - 1) ? 1 : 0);
        }
        cur = out;
    }

    // final projection: d_out = cur @ W_out + b_out  (Hc = 64, NT = 1)
    wt_kernel<<<(D_H * D_OUT + 255) / 256, 256, 0, stream>>>(Wout, wt_hi[0], wt_lo[0], D_H, D_OUT);
    dim3 gridF(N_NODES / 16, 1);
    gemm_wmma_kernel<1><<<gridF, 128, 0, stream>>>(
        cur, nullptr, wt_hi[0], wt_lo[0], bout, (float*)d_out, D_H, D_OUT, 0, 0);
}
